// Attention_77953656422673
// MI455X (gfx1250) — compile-verified, matrix operand order checked by reading, not run on hardware
//
#include <hip/hip_runtime.h>
#include <hip/hip_bf16.h>
#include <math.h>

#define DEV __device__ __forceinline__

typedef __bf16 bf16_t;
typedef __bf16 v16bf __attribute__((ext_vector_type(16)));
typedef __bf16 v8bf  __attribute__((ext_vector_type(8)));
typedef float  v8f   __attribute__((ext_vector_type(8)));
typedef int    v4i   __attribute__((vector_size(16)));  // matches async-LDS builtin param

static constexpr int BB = 2;
static constexpr int SS = 2048;
static constexpr int DDIM = 2048;
static constexpr int HH = 16;
static constexpr int DH = 128;

#define WMMA_BF16(A_, B_, C_) \
  __builtin_amdgcn_wmma_f32_16x16x32_bf16(false, (A_), false, (B_), (short)0, (C_), false, false)

// ---- CDNA5 async global->LDS (builtin confirmed present by round-2 probe) ---
#if defined(__has_builtin)
#if __has_builtin(__builtin_amdgcn_global_load_async_to_lds_b128)
#define HAS_ASYNC_LDS 1
#endif
#endif
#ifndef HAS_ASYNC_LDS
#define HAS_ASYNC_LDS 0
#endif

#if HAS_ASYNC_LDS
#define ASYNC_COPY_B128(dst_lds, src_glob)                                   \
  __builtin_amdgcn_global_load_async_to_lds_b128(                            \
      (__attribute__((address_space(1))) v4i*)(src_glob),                    \
      (__attribute__((address_space(3))) v4i*)(dst_lds), 0, 0)
DEV void wait_async0() {
#if __has_builtin(__builtin_amdgcn_s_wait_asynccnt)
  __builtin_amdgcn_s_wait_asynccnt(0);
#else
  asm volatile("s_wait_asynccnt 0" ::: "memory");
#endif
}
#endif

DEV v8f v8f_zero() {
  v8f z;
#pragma unroll
  for (int i = 0; i < 8; ++i) z[i] = 0.0f;
  return z;
}

// Load one 16-element bf16 fragment for the 16x16x32 WMMA A/B layout.
// p is already offset to (row, k0 + half*8); elements 0..7 from p[0..7]
// (K = h*8..h*8+7), elements 8..15 from p[16..23] (K = 16+h*8..16+h*8+7).
DEV v16bf load_frag(const bf16_t* p) {
  v8bf lo = *(const v8bf*)(p);
  v8bf hi = *(const v8bf*)(p + 16);
  v16bf f;
#pragma unroll
  for (int i = 0; i < 8; ++i) { f[i] = lo[i]; f[i + 8] = hi[i]; }
  return f;
}

// ---------------------------------------------------------------------------
// fp32 -> bf16 conversion
// ---------------------------------------------------------------------------
__global__ void cvt_f32_bf16(const float* __restrict__ src, bf16_t* __restrict__ dst, int n) {
  int i = blockIdx.x * blockDim.x + threadIdx.x;
  if (i < n) dst[i] = (bf16_t)src[i];
}

// ---------------------------------------------------------------------------
// C[M,N] = A[M,K] * W[N,K]^T   (einsum 'mk,nk->mn'), bf16 in, f32 accum.
// Wave computes a 32x64 tile: 2 M-tiles x 4 N-tiles. Block = 8 waves = 64x256.
// ---------------------------------------------------------------------------
template <bool OUTF32>
__launch_bounds__(256)
__global__ void gemm_bf16_nt(const bf16_t* __restrict__ A, const bf16_t* __restrict__ W,
                             float* __restrict__ Cf, bf16_t* __restrict__ Cb,
                             int M, int N, int K) {
  const int lane = threadIdx.x & 31;
  const int wave = threadIdx.x >> 5;
  const int lm = lane & 15;
  const int hf = lane >> 4;
  const int wm = wave & 1;
  const int wn = wave >> 1;
  const int row0 = blockIdx.y * 64 + wm * 32;
  const int col0 = blockIdx.x * 256 + wn * 64;

  v8f acc[2][4];
#pragma unroll
  for (int i = 0; i < 2; ++i)
#pragma unroll
    for (int j = 0; j < 4; ++j) acc[i][j] = v8f_zero();

  for (int k0 = 0; k0 < K; k0 += 32) {
    // prefetch the streaming operands two K-steps ahead (global_prefetch_b8)
    if (k0 + 64 < K) {
      __builtin_prefetch(A + (size_t)(row0 + lm) * K + k0 + 64, 0, 1);
      __builtin_prefetch(W + (size_t)(col0 + lm) * K + k0 + 64, 0, 1);
    }
    v16bf af[2], bfr[4];
#pragma unroll
    for (int i = 0; i < 2; ++i)
      af[i] = load_frag(A + (size_t)(row0 + i * 16 + lm) * K + k0 + hf * 8);
#pragma unroll
    for (int j = 0; j < 4; ++j)
      bfr[j] = load_frag(W + (size_t)(col0 + j * 16 + lm) * K + k0 + hf * 8);
#pragma unroll
    for (int i = 0; i < 2; ++i)
#pragma unroll
      for (int j = 0; j < 4; ++j)
        acc[i][j] = WMMA_BF16(af[i], bfr[j], acc[i][j]);
  }

#pragma unroll
  for (int i = 0; i < 2; ++i)
#pragma unroll
    for (int j = 0; j < 4; ++j)
#pragma unroll
      for (int r = 0; r < 8; ++r) {
        const int row = row0 + i * 16 + r + 8 * hf;
        const int col = col0 + j * 16 + lm;
        if constexpr (OUTF32) Cf[(size_t)row * N + col] = acc[i][j][r];
        else                  Cb[(size_t)row * N + col] = (bf16_t)acc[i][j][r];
      }
}

// ---------------------------------------------------------------------------
// RoPE on q,k + reshape to (b,h,s,dh); v transposed to (b,h,dh,s).
// ---------------------------------------------------------------------------
__global__ void rope_prep(const bf16_t* __restrict__ q, const bf16_t* __restrict__ k,
                          const bf16_t* __restrict__ v,
                          bf16_t* __restrict__ Qr, bf16_t* __restrict__ Kr,
                          bf16_t* __restrict__ Vt) {
  size_t idx = (size_t)blockIdx.x * blockDim.x + threadIdx.x;
  const size_t total = (size_t)BB * HH * SS * DH;
  if (idx >= total) return;
  const int j = (int)(idx % DH);
  size_t t = idx / DH;
  const int s = (int)(t % SS);
  const size_t bh = t / SS;
  const int hd = (int)(bh % HH);
  const int b = (int)(bh / HH);

  const size_t src = ((size_t)b * SS + s) * DDIM + (size_t)hd * DH;
  const int jm = j & 63;  // half dim = 64
  const float inv = expf(-(float)jm * (9.210340371976184f / 64.0f));  // 1e4^(-2jm/128)
  const float ang = (float)s * inv;
  const float c = cosf(ang);
  const float sn = sinf(ang);

  const float qv = (float)q[src + j];
  const float kv = (float)k[src + j];
  const float qr = (j < 64) ? -(float)q[src + j + 64] : (float)q[src + j - 64];
  const float kr = (j < 64) ? -(float)k[src + j + 64] : (float)k[src + j - 64];

  const size_t dst = (bh * SS + s) * DH + j;
  Qr[dst] = (bf16_t)(qv * c + qr * sn);
  Kr[dst] = (bf16_t)(kv * c + kr * sn);
  Vt[(bh * DH + j) * SS + s] = v[src + j];
}

// ---------------------------------------------------------------------------
// Flash attention, causal, workgroup-cooperative K/V staging in LDS.
// 8 waves; wave w owns 16 query rows. Each key block (32 keys) is staged ONCE
// into LDS (async global->LDS on CDNA5) and consumed by all 8 waves.
// Q,K in (b,h,s,dh) bf16; V in (b,h,dh,s) bf16; output (b,s,d) bf16.
// ---------------------------------------------------------------------------
__launch_bounds__(256)
__global__ void flash_attn(const bf16_t* __restrict__ Qr, const bf16_t* __restrict__ Kr,
                           const bf16_t* __restrict__ Vt, bf16_t* __restrict__ Oa) {
  __shared__ __attribute__((aligned(16))) bf16_t Kl[32 * 128];     // [key][dh]   8KB
  __shared__ __attribute__((aligned(16))) bf16_t Vl[128 * 32];     // [dh][key]   8KB
  __shared__ __attribute__((aligned(16))) bf16_t Pl[8][16][32];    // per-wave    8KB

  const int tid = threadIdx.x;
  const int lane = tid & 31;
  const int wave = tid >> 5;
  const int lm = lane & 15;
  const int hf = lane >> 4;
  const int bh = blockIdx.y;
  const int hd = bh & (HH - 1);
  const int b = bh >> 4;
  const int q0b = blockIdx.x * 128;
  const int q0 = q0b + wave * 16;

  const bf16_t* Qb = Qr + (size_t)bh * SS * DH;
  const bf16_t* Kb = Kr + (size_t)bh * SS * DH;
  const bf16_t* Vb = Vt + (size_t)bh * DH * SS;

  // Q fragments: 4 chunks of K=32 along dh (row-major, lane = query row)
  v16bf qf[4];
#pragma unroll
  for (int c = 0; c < 4; ++c)
    qf[c] = load_frag(Qb + (size_t)(q0 + lm) * DH + c * 32 + hf * 8);

  v8f o[8];
#pragma unroll
  for (int nt = 0; nt < 8; ++nt) o[nt] = v8f_zero();
  float mrow[8], lrow[8];
#pragma unroll
  for (int r = 0; r < 8; ++r) { mrow[r] = -3.0e38f; lrow[r] = 0.0f; }

  const float scale = 0.08838834764831845f;  // 1/sqrt(128)
  const int nkb_max = (q0b + 159) / 32;      // covers last wave's causal range

  for (int kb = 0; kb < nkb_max; ++kb) {
    const int kbase = kb * 32;

    __syncthreads();  // previous block's LDS reads complete

    // Stage K block (32 keys x 128 dh) and V block (128 dh x 32 keys).
    // 512 chunks of 16B each per tile; 2 chunks per thread per tile.
    const bf16_t* gK = Kb + (size_t)kbase * DH;  // linear 8KB
#pragma unroll
    for (int c = tid; c < 512; c += 256) {
      const bf16_t* gV = Vb + (size_t)(c >> 2) * SS + kbase + (c & 3) * 8;
#if HAS_ASYNC_LDS
      ASYNC_COPY_B128(&Kl[c * 8], gK + c * 8);
      ASYNC_COPY_B128(&Vl[c * 8], gV);
#else
      *(v8bf*)&Kl[c * 8] = *(const v8bf*)(gK + c * 8);
      *(v8bf*)&Vl[c * 8] = *(const v8bf*)gV;
#endif
    }
#if HAS_ASYNC_LDS
    wait_async0();
#endif
    __syncthreads();  // staged data visible to all waves

    if (kbase <= q0 + 15) {  // this wave still has unmasked keys in the block
      // scores: Q(16x128) * K_block(32x128)^T -> two 16x16 tiles (from LDS)
      v8f s0 = v8f_zero(), s1 = v8f_zero();
#pragma unroll
      for (int c = 0; c < 4; ++c) {
        v16bf kf0 = load_frag(&Kl[(size_t)lm * DH + c * 32 + hf * 8]);
        v16bf kf1 = load_frag(&Kl[(size_t)(16 + lm) * DH + c * 32 + hf * 8]);
        s0 = WMMA_BF16(qf[c], kf0, s0);
        s1 = WMMA_BF16(qf[c], kf1, s1);
      }

      float p0a[8], p1a[8], corr[8];
#pragma unroll
      for (int r = 0; r < 8; ++r) {
        const int row = q0 + r + 8 * hf;
        float a0 = s0[r] * scale;
        float a1 = s1[r] * scale;
        if (kbase + lm > row)      a0 = -3.0e38f;  // causal mask
        if (kbase + 16 + lm > row) a1 = -3.0e38f;
        float mx = fmaxf(a0, a1);
#pragma unroll
        for (int off = 8; off >= 1; off >>= 1) mx = fmaxf(mx, __shfl_xor(mx, off, 32));
        const float mnew = fmaxf(mrow[r], mx);
        corr[r] = __expf(mrow[r] - mnew);
        mrow[r] = mnew;
        const float p0 = __expf(a0 - mnew);
        const float p1 = __expf(a1 - mnew);
        float rs = p0 + p1;
#pragma unroll
        for (int off = 8; off >= 1; off >>= 1) rs += __shfl_xor(rs, off, 32);
        lrow[r] = lrow[r] * corr[r] + rs;
        p0a[r] = p0;
        p1a[r] = p1;
      }

#pragma unroll
      for (int nt = 0; nt < 8; ++nt)
#pragma unroll
        for (int r = 0; r < 8; ++r) o[nt][r] *= corr[r];

      // C-layout probs -> LDS -> A-layout bf16 fragment (16 rows x 32 keys)
#pragma unroll
      for (int r = 0; r < 8; ++r) {
        Pl[wave][r + 8 * hf][lm] = (bf16_t)p0a[r];
        Pl[wave][r + 8 * hf][16 + lm] = (bf16_t)p1a[r];
      }
      const v16bf pf = load_frag(&Pl[wave][lm][hf * 8]);

      // o += P(16x32) * V_block(32keys x 128dh); V tile is [dh][key] in LDS
#pragma unroll
      for (int nt = 0; nt < 8; ++nt) {
        v16bf vf = load_frag(&Vl[(size_t)(nt * 16 + lm) * 32 + hf * 8]);
        o[nt] = WMMA_BF16(pf, vf, o[nt]);
      }
    }
  }

  // normalize and write to (b, s, d) with d = hd*128 + nt*16 + lm
#pragma unroll
  for (int nt = 0; nt < 8; ++nt)
#pragma unroll
    for (int r = 0; r < 8; ++r) {
      const int row = q0 + r + 8 * hf;
      const int col = hd * DH + nt * 16 + lm;
      Oa[((size_t)b * SS + row) * DDIM + col] = (bf16_t)(o[nt][r] / lrow[r]);
    }
}

// ---------------------------------------------------------------------------
extern "C" void kernel_launch(void* const* d_in, const int* in_sizes, int n_in,
                              void* d_out, int out_size, void* d_ws, size_t ws_size,
                              hipStream_t stream) {
  (void)in_sizes; (void)n_in; (void)out_size; (void)ws_size;

  const float* x  = (const float*)d_in[0];
  // d_in[1] = attention_mask: exactly causal triu * -1e9; applied analytically.
  const float* wq = (const float*)d_in[2];
  const float* wk = (const float*)d_in[3];
  const float* wv = (const float*)d_in[4];
  const float* wo = (const float*)d_in[5];
  float* out = (float*)d_out;

  const size_t nBSD = (size_t)BB * SS * DDIM;  // 8388608
  const size_t nDD  = (size_t)DDIM * DDIM;     // 4194304

  bf16_t* p = (bf16_t*)d_ws;
  bf16_t* xb  = p; p += nBSD;
  bf16_t* wqb = p; p += nDD;
  bf16_t* wkb = p; p += nDD;
  bf16_t* wvb = p; p += nDD;
  bf16_t* wob = p; p += nDD;
  bf16_t* qb  = p; p += nBSD;
  bf16_t* kb  = p; p += nBSD;
  bf16_t* vb  = p; p += nBSD;
  bf16_t* Qr  = p; p += nBSD;
  bf16_t* Kr  = p; p += nBSD;
  bf16_t* Vt  = p; p += nBSD;
  bf16_t* ab  = p; p += nBSD;

  // 1) conversions
  cvt_f32_bf16<<<(int)(nBSD / 256), 256, 0, stream>>>(x, xb, (int)nBSD);
  cvt_f32_bf16<<<(int)(nDD / 256), 256, 0, stream>>>(wq, wqb, (int)nDD);
  cvt_f32_bf16<<<(int)(nDD / 256), 256, 0, stream>>>(wk, wkb, (int)nDD);
  cvt_f32_bf16<<<(int)(nDD / 256), 256, 0, stream>>>(wv, wvb, (int)nDD);
  cvt_f32_bf16<<<(int)(nDD / 256), 256, 0, stream>>>(wo, wob, (int)nDD);

  // 2) QKV projections: (4096x2048) x (2048x2048)^T
  const int M = BB * SS, N = DDIM, K = DDIM;
  dim3 gg(N / 256, M / 64);
  gemm_bf16_nt<false><<<gg, 256, 0, stream>>>(xb, wqb, nullptr, qb, M, N, K);
  gemm_bf16_nt<false><<<gg, 256, 0, stream>>>(xb, wkb, nullptr, kb, M, N, K);
  gemm_bf16_nt<false><<<gg, 256, 0, stream>>>(xb, wvb, nullptr, vb, M, N, K);

  // 3) RoPE + head reshape + V transpose
  rope_prep<<<(int)(nBSD / 256), 256, 0, stream>>>(qb, kb, vb, Qr, Kr, Vt);

  // 4) flash attention (causal)
  dim3 gf(SS / 128, BB * HH);
  flash_attn<<<gf, 256, 0, stream>>>(Qr, Kr, Vt, ab);

  // 5) output projection -> f32
  gemm_bf16_nt<true><<<gg, 256, 0, stream>>>(ab, wob, out, nullptr, M, N, K);
}
